// SlidingWindowAttention_49941879718134
// MI455X (gfx1250) — matrix operand order repaired and checked by reading, not compiled, not run on hardware
//
#include <hip/hip_runtime.h>

// ---------------------------------------------------------------------------
// Sliding-window attention for MI455X (gfx1250).
// Roofline: ~26 GFLOP total vs ~40MB of HBM traffic (1.7us @ 23.3TB/s); the
// ~30MB working set is L2-resident (192MB) -> matrix-op bound. So: bf16
// WMMA (v_wmma_f32_16x16x32_bf16, fp32 accumulate) for QKV / QK^T / PV / Wo,
// and the Tensor Data Mover to stage the block-shared weight panel in LDS
// (it is read by all 4 waves of a workgroup; TDM removes the 4x duplicate
// L2 reads and overlaps the copy with WMMA via TENSORcnt double-buffering).
// ---------------------------------------------------------------------------

#define HN    8
#define DD    64
#define NSEQ  4096
#define HID   1024
#define INNER 512
#define WIN   512

typedef __bf16 bf16;
typedef __attribute__((ext_vector_type(8)))  bf16  bf16x8;
typedef __attribute__((ext_vector_type(16))) bf16  bf16x16;
typedef __attribute__((ext_vector_type(8)))  float f32x8;
typedef __attribute__((ext_vector_type(4)))  unsigned int u32x4;
typedef __attribute__((ext_vector_type(8)))  int   i32x8;
typedef __attribute__((ext_vector_type(4)))  int   i32x4;

#if defined(__has_builtin)
#if __has_builtin(__builtin_amdgcn_tensor_load_to_lds) && \
    __has_builtin(__builtin_amdgcn_s_wait_tensorcnt)
#define HAVE_TDM 1
#endif
#endif
#ifndef HAVE_TDM
#define HAVE_TDM 0
#endif

union BF16V { bf16x16 v; bf16x8 h[2]; };

static __device__ inline f32x8 zero8() {
  f32x8 z = {0.f,0.f,0.f,0.f,0.f,0.f,0.f,0.f};
  return z;
}

static __device__ inline f32x8 wmma_bf16(bf16x16 a, bf16x16 b, f32x8 c) {
  // 8 args: (neg_a, A, neg_b, B, c_mod, C, reuse_a, reuse_b)
  return __builtin_amdgcn_wmma_f32_16x16x32_bf16(false, a, false, b,
                                                 (short)0, c, false, false);
}

// A-frag (16x32, 16-bit): lane m=L&15, half=L>>4 ; element j holds
// K = 16*half + j  -> one contiguous 32B run.
static __device__ inline bf16x16 load_a_frag(const bf16* rowp, int half) {
  BF16V r;
  r.h[0] = *reinterpret_cast<const bf16x8*>(rowp + 16 * half);
  r.h[1] = *reinterpret_cast<const bf16x8*>(rowp + 16 * half + 8);
  return r.v;
}

// B-frag (32x16, 16-bit): lane n=L&15, half=L>>4 ; element j holds
// K = 16*half + j. Caller passes pointer already offset by 16*half along K.
static __device__ inline bf16x16 load_b_frag(const bf16* kp) {
  BF16V r;
  r.h[0] = *reinterpret_cast<const bf16x8*>(kp);
  r.h[1] = *reinterpret_cast<const bf16x8*>(kp + 8);
  return r.v;
}

// ---------------------------------------------------------------------------
// K0: fp32 -> bf16 conversion
// ---------------------------------------------------------------------------
__global__ void k_cvt_bf16(const float* __restrict__ src,
                           bf16* __restrict__ dst, int n) {
  int i = blockIdx.x * blockDim.x + threadIdx.x;
  if (i < n) dst[i] = (bf16)src[i];
}

// ---------------------------------------------------------------------------
// K1: sem = cumsum(active) - 1 ; single block of 1024 threads, 4 elems each
// ---------------------------------------------------------------------------
__global__ __launch_bounds__(1024)
void k_sem_scan(const unsigned char* __restrict__ amask,
                int* __restrict__ sem, int* __restrict__ act) {
  __shared__ int tot[1024];
  int t = threadIdx.x;
  int base = t * 4;
  int loc[4];
  int s = 0;
#pragma unroll
  for (int j = 0; j < 4; ++j) {
    int a = amask[base + j] ? 1 : 0;
    act[base + j] = a;
    s += a;
    loc[j] = s;
  }
  tot[t] = s;
  __syncthreads();
  for (int off = 1; off < 1024; off <<= 1) {
    int v = (t >= off) ? tot[t - off] : 0;
    __syncthreads();
    tot[t] += v;
    __syncthreads();
  }
  int excl = tot[t] - s;
#pragma unroll
  for (int j = 0; j < 4; ++j) sem[base + j] = excl + loc[j] - 1;
}

// ---------------------------------------------------------------------------
// K2: fused QKV GEMM (M=4096, N=512, K=1024) + RoPE, bf16 WMMA.
// Per wave: 32 rows x 64 cols (= one head's features) -> 2x4 accum frags.
// grid = (32, 8 heads, 3 mats), block = 128 (4 waves over row tiles).
// The 64x128 B chunk is shared by all 4 waves -> TDM-staged into LDS with
// double buffering (TENSORcnt). Padded LDS rows (272B) avoid bank conflicts.
// q,k stored [h][n][d]; v stored transposed [h][d][n] for PV B-frags.
// ---------------------------------------------------------------------------
#define KCHUNK 128
#define BROW   136   // 128 elems + TDM pad (4 DWORDs = 8 elems) = 272 bytes

#if HAVE_TDM
// Issue one TDM 2D-tile load: 64 rows (features) x 128 K bf16 from W into LDS.
// D# group0: count=1 | lds_addr | global_addr | type=2
// D# group1: data_size=2B, pad every 64 DWORDs by 4 DWORDs,
//            tensor = 512x1024 (stride 1024), tile = 64x128.
static __device__ inline void tdm_load_b_tile(const bf16* Wbase, int h, int kk,
                                              unsigned ldsaddr) {
  unsigned long long ga =
      (unsigned long long)Wbase +
      2ull * ((unsigned long long)h * DD * HID + (unsigned)kk);
  u32x4 g0;
  g0[0] = 1u;                                         // count=1 (user D#)
  g0[1] = ldsaddr;                                    // lds_addr (bytes)
  g0[2] = (unsigned)ga;                               // global_addr[31:0]
  g0[3] = (unsigned)((ga >> 32) & 0x01FFFFFFull) | (2u << 30);  // addr|type=2
  i32x8 g1;
  g1[0] = (int)((1u << 16) |   // data_size = 1 -> 2 bytes
                (1u << 20) |   // pad_enable
                (5u << 22) |   // pad_interval: 64 DWORDs (one 128-elem row)
                (3u << 25));   // pad_amount: 4 DWORDs (8 bf16)
  g1[1] = (int)((HID & 0xFFFFu) << 16);               // tensor_dim0 lo16
  g1[2] = (int)((HID >> 16) | ((INNER & 0xFFFFu) << 16));  // dim0 hi | dim1 lo
  g1[3] = (int)((INNER >> 16) | ((unsigned)KCHUNK << 16)); // dim1 hi | tile0
  g1[4] = (int)DD;                                    // tile_dim1=64, tile2=0
  g1[5] = (int)HID;                                   // dim0_stride[31:0]
  g1[6] = 0;                                          // stride hi | dim1_stride
  g1[7] = 0;
  i32x4 gz = {0, 0, 0, 0};
#if __clang_major__ >= 23
  i32x8 gz8 = {0, 0, 0, 0, 0, 0, 0, 0};
  __builtin_amdgcn_tensor_load_to_lds(g0, g1, gz, gz, gz8, 0);
#else
  __builtin_amdgcn_tensor_load_to_lds(g0, g1, gz, gz, 0);
#endif
}
#endif  // HAVE_TDM

__global__ __launch_bounds__(128)
void k_qkv_rope(const bf16* __restrict__ xb,
                const bf16* __restrict__ wq, const bf16* __restrict__ wk,
                const bf16* __restrict__ wv, const int* __restrict__ pid,
                bf16* __restrict__ qh, bf16* __restrict__ kh,
                bf16* __restrict__ vt) {
#if HAVE_TDM
  __shared__ bf16 bstage[2][DD * BROW];   // 2 x 17408 B, double-buffered
#endif
  const int wave = threadIdx.x >> 5;
  const int lane = threadIdx.x & 31;
  const int lm = lane & 15, half = lane >> 4;
  const int n0 = (blockIdx.x * 4 + wave) * 32;
  const int h  = blockIdx.y;
  const int z  = blockIdx.z;
  const bf16* W = (z == 0) ? wq : ((z == 1) ? wk : wv);

  f32x8 acc[2][4];
#pragma unroll
  for (int rf = 0; rf < 2; ++rf)
#pragma unroll
    for (int c = 0; c < 4; ++c) acc[rf][c] = zero8();

  const bf16* arow0 = xb + (size_t)(n0 + lm) * HID;
  const bf16* arow1 = xb + (size_t)(n0 + 16 + lm) * HID;

#if HAVE_TDM
  const unsigned ldsa0 = (unsigned)(unsigned long long)&bstage[0][0];
  const unsigned ldsa1 = (unsigned)(unsigned long long)&bstage[1][0];
  if (wave == 0) tdm_load_b_tile(W, h, 0, ldsa0);
  int cur = 0;
  for (int kk = 0; kk < HID; kk += KCHUNK) {
    if (wave == 0) __builtin_amdgcn_s_wait_tensorcnt(0);
    __syncthreads();  // chunk `kk` resident; prior chunk's readers all done
    if (wave == 0 && kk + KCHUNK < HID)
      tdm_load_b_tile(W, h, kk + KCHUNK, (cur == 0) ? ldsa1 : ldsa0);
    const bf16* Bs = &bstage[cur][0];
#pragma unroll
    for (int k2 = 0; k2 < KCHUNK; k2 += 32) {
      __builtin_prefetch(arow0 + kk + k2 + 256, 0, 1);  // global_prefetch_b8
      bf16x16 a0 = load_a_frag(arow0 + kk + k2, half);
      bf16x16 a1 = load_a_frag(arow1 + kk + k2, half);
      // Hoist all 4 B-frag DS loads so the scheduler can batch ds_load_b128
      // and use staged s_wait_dscnt instead of a full stop per WMMA pair.
      bf16x16 bfr[4];
#pragma unroll
      for (int c = 0; c < 4; ++c)
        bfr[c] = load_b_frag(Bs + (c * 16 + lm) * BROW + k2 + 16 * half);
#pragma unroll
      for (int c = 0; c < 4; ++c) {
        acc[0][c] = wmma_bf16(a0, bfr[c], acc[0][c]);
        acc[1][c] = wmma_bf16(a1, bfr[c], acc[1][c]);
      }
    }
    cur ^= 1;
  }
#else
  for (int kk = 0; kk < HID; kk += 32) {
    __builtin_prefetch(arow0 + kk + 256, 0, 1);
    bf16x16 a0 = load_a_frag(arow0 + kk, half);
    bf16x16 a1 = load_a_frag(arow1 + kk, half);
    bf16x16 bfr[4];
#pragma unroll
    for (int c = 0; c < 4; ++c)
      bfr[c] = load_b_frag(W + (size_t)(h * DD + c * 16 + lm) * HID + kk +
                           16 * half);
#pragma unroll
    for (int c = 0; c < 4; ++c) {
      acc[0][c] = wmma_bf16(a0, bfr[c], acc[0][c]);
      acc[1][c] = wmma_bf16(a1, bfr[c], acc[1][c]);
    }
  }
#endif

  if (z < 2) {
    // RoPE: pair (d, d+32) lives in frags (c, c+2) of the SAME lane.
    bf16* dst = (z == 0) ? qh : kh;
#pragma unroll
    for (int rf = 0; rf < 2; ++rf) {
#pragma unroll
      for (int c = 0; c < 2; ++c) {
        const int d_lo = c * 16 + lm;                  // 0..31
        const float invf = __expf((float)d_lo * (-2.0f / 64.0f) * 9.210340372f);
#pragma unroll
        for (int r = 0; r < 8; ++r) {
          const int row = n0 + rf * 16 + r + 8 * half;
          const float pos = (float)pid[row];
          const float ang = pos * invf;
          const float cs = __cosf(ang), sn = __sinf(ang);
          const float lo = acc[rf][c][r];
          const float hi = acc[rf][c + 2][r];
          const size_t base = ((size_t)h * NSEQ + row) * DD;
          dst[base + d_lo]      = (bf16)(lo * cs - hi * sn);
          dst[base + d_lo + 32] = (bf16)(hi * cs + lo * sn);
        }
      }
    }
  } else {
    // V transposed: vt[h][d][n]; 8 accumulator rows are contiguous n -> b128.
#pragma unroll
    for (int rf = 0; rf < 2; ++rf) {
#pragma unroll
      for (int c = 0; c < 4; ++c) {
        bf16x8 pk;
#pragma unroll
        for (int r = 0; r < 8; ++r) pk[r] = (bf16)acc[rf][c][r];
        const int d = c * 16 + lm;
        const int n = n0 + rf * 16 + 8 * half;
        *reinterpret_cast<bf16x8*>(vt + ((size_t)h * DD + d) * NSEQ + n) = pk;
      }
    }
  }
}

// ---------------------------------------------------------------------------
// K3: flash-style sliding-window attention. One wave = (head, 16-query tile).
// Streams 32-key chunks; skips chunks entirely outside the semantic window.
// P converted C-layout -> A-layout via a per-wave LDS slab (DS_ path).
// ---------------------------------------------------------------------------
#define WPB 4
__global__ __launch_bounds__(32 * WPB)
void k_attn(const bf16* __restrict__ qh, const bf16* __restrict__ kh,
            const bf16* __restrict__ vt, const int* __restrict__ sem,
            const int* __restrict__ act, bf16* __restrict__ attnb) {
  __shared__ bf16 plds[WPB][16 * 32];
  const int wave = threadIdx.x >> 5;
  const int lane = threadIdx.x & 31;
  const int lm = lane & 15, half = lane >> 4;
  const int gw = blockIdx.x * WPB + wave;  // 0..2047
  const int h  = gw >> 8;                  // 8 heads
  const int qt = gw & 255;                 // 256 query tiles of 16
  const int qn0 = qt * 16;
  const int qlast = qn0 + 15;

  const bf16* Q = qh + (size_t)h * NSEQ * DD;
  const bf16* K = kh + (size_t)h * NSEQ * DD;
  const bf16* V = vt + (size_t)h * DD * NSEQ;

  // Q fragments for the whole tile (d 0..31 and 32..63)
  const bf16x16 aq0 = load_a_frag(Q + (size_t)(qn0 + lm) * DD, half);
  const bf16x16 aq1 = load_a_frag(Q + (size_t)(qn0 + lm) * DD + 32, half);

  int qsem[8], qact[8];
#pragma unroll
  for (int r = 0; r < 8; ++r) {
    const int row = qn0 + r + 8 * half;
    qsem[r] = sem[row];
    qact[r] = act[row];
  }
  const int qsem_min = sem[qn0];  // sem is non-decreasing

  float Mrow[8], Lrow[8];
#pragma unroll
  for (int r = 0; r < 8; ++r) { Mrow[r] = -3.0e38f; Lrow[r] = 0.f; }
  f32x8 o[4];
#pragma unroll
  for (int c = 0; c < 4; ++c) o[c] = zero8();

  const float scale = 0.125f;  // 1/sqrt(64)
  bf16* pw = &plds[wave][0];

  for (int kt = 0; kt <= qlast; kt += 32) {
    const int klast = (kt + 31 < qlast) ? (kt + 31) : qlast;
    if (sem[klast] + WIN <= qsem_min) continue;  // wave-uniform skip

    // ---- S = Q K^T for 32 keys (two 16-key column frags) ----
    // Hoist the 4 K-matrix B-frag loads so vmem issues batch up front.
    bf16x16 bk[4];
    bk[0] = load_b_frag(K + (size_t)(kt + lm) * DD + 16 * half);
    bk[1] = load_b_frag(K + (size_t)(kt + lm) * DD + 32 + 16 * half);
    bk[2] = load_b_frag(K + (size_t)(kt + 16 + lm) * DD + 16 * half);
    bk[3] = load_b_frag(K + (size_t)(kt + 16 + lm) * DD + 32 + 16 * half);
    f32x8 s0 = zero8(), s1 = zero8();
    s0 = wmma_bf16(aq0, bk[0], s0);
    s0 = wmma_bf16(aq1, bk[1], s0);
    s1 = wmma_bf16(aq0, bk[2], s1);
    s1 = wmma_bf16(aq1, bk[3], s1);

    const int k0i = kt + lm, k1i = kt + 16 + lm;
    const int ks0 = sem[k0i], ks1 = sem[k1i];
    const int ka0 = act[k0i], ka1 = act[k1i];

    // ---- masked online softmax (row = r + 8*half, spread over 16 lanes) ----
#pragma unroll
    for (int r = 0; r < 8; ++r) {
      const int qs = qsem[r];
      const bool qa = (qact[r] != 0);
      const bool m0 = qa && ka0 && (ks0 <= qs) && (qs - ks0 < WIN);
      const bool m1 = qa && ka1 && (ks1 <= qs) && (qs - ks1 < WIN);
      const float v0 = m0 ? s0[r] * scale : -3.0e38f;
      const float v1 = m1 ? s1[r] * scale : -3.0e38f;

      float cmax = fmaxf(v0, v1);
#pragma unroll
      for (int mk = 1; mk < 16; mk <<= 1)
        cmax = fmaxf(cmax, __shfl_xor(cmax, mk, 32));

      const float Mn = fmaxf(Mrow[r], cmax);
      const float alpha = __expf(Mrow[r] - Mn);
      const float p0 = m0 ? __expf(v0 - Mn) : 0.f;
      const float p1 = m1 ? __expf(v1 - Mn) : 0.f;

      float rs = p0 + p1;
#pragma unroll
      for (int mk = 1; mk < 16; mk <<= 1) rs += __shfl_xor(rs, mk, 32);

      Lrow[r] = Lrow[r] * alpha + rs;
      Mrow[r] = Mn;
#pragma unroll
      for (int c = 0; c < 4; ++c) o[c][r] *= alpha;

      const int m = r + 8 * half;
      pw[m * 32 + lm]      = (bf16)p0;   // DS stores
      pw[m * 32 + 16 + lm] = (bf16)p1;
    }

    // ---- reload P in A-layout from LDS (two ds_load_b128 per lane) ----
    BF16V pa;
    pa.h[0] = *reinterpret_cast<const bf16x8*>(pw + lm * 32 + 16 * half);
    pa.h[1] = *reinterpret_cast<const bf16x8*>(pw + lm * 32 + 16 * half + 8);

    // ---- O += P (16x32) * V (32x64), V frags batched ----
    bf16x16 bv[4];
#pragma unroll
    for (int c = 0; c < 4; ++c)
      bv[c] = load_b_frag(V + (size_t)(c * 16 + lm) * NSEQ + kt + 16 * half);
#pragma unroll
    for (int c = 0; c < 4; ++c) o[c] = wmma_bf16(pa.v, bv[c], o[c]);
  }

  // ---- finalize & store attn output bf16 [n][H*D] ----
#pragma unroll
  for (int r = 0; r < 8; ++r) {
    const float inv = (Lrow[r] > 0.f) ? (1.0f / Lrow[r]) : 0.f;
    const int row = qn0 + r + 8 * half;
#pragma unroll
    for (int c = 0; c < 4; ++c)
      attnb[(size_t)row * INNER + h * DD + c * 16 + lm] =
          (bf16)(o[c][r] * inv);
  }
}

// ---------------------------------------------------------------------------
// K4: out = attn @ Wo^T (M=4096, N=1024, K=512). 32x32 tile per wave.
// grid = (32, 32), block = 128.
// ---------------------------------------------------------------------------
__global__ __launch_bounds__(128)
void k_out_proj(const bf16* __restrict__ attnb, const bf16* __restrict__ wo,
                float* __restrict__ out) {
  const int wave = threadIdx.x >> 5;
  const int lane = threadIdx.x & 31;
  const int lm = lane & 15, half = lane >> 4;
  const int n0 = (blockIdx.x * 4 + wave) * 32;
  const int c0 = blockIdx.y * 32;

  f32x8 acc[2][2];
#pragma unroll
  for (int rf = 0; rf < 2; ++rf)
#pragma unroll
    for (int cf = 0; cf < 2; ++cf) acc[rf][cf] = zero8();

  const bf16* arow0 = attnb + (size_t)(n0 + lm) * INNER;
  const bf16* arow1 = attnb + (size_t)(n0 + 16 + lm) * INNER;

  for (int kk = 0; kk < INNER; kk += 32) {
    bf16x16 a0 = load_a_frag(arow0 + kk, half);
    bf16x16 a1 = load_a_frag(arow1 + kk, half);
    bf16x16 b0 =
        load_b_frag(wo + (size_t)(c0 + lm) * INNER + kk + 16 * half);
    bf16x16 b1 =
        load_b_frag(wo + (size_t)(c0 + 16 + lm) * INNER + kk + 16 * half);
    acc[0][0] = wmma_bf16(a0, b0, acc[0][0]);
    acc[0][1] = wmma_bf16(a0, b1, acc[0][1]);
    acc[1][0] = wmma_bf16(a1, b0, acc[1][0]);
    acc[1][1] = wmma_bf16(a1, b1, acc[1][1]);
  }

#pragma unroll
  for (int rf = 0; rf < 2; ++rf)
#pragma unroll
    for (int cf = 0; cf < 2; ++cf)
#pragma unroll
      for (int r = 0; r < 8; ++r) {
        const int row = n0 + rf * 16 + r + 8 * half;
        const int col = c0 + cf * 16 + lm;
        out[(size_t)row * HID + col] = acc[rf][cf][r];
      }
}

// ---------------------------------------------------------------------------
extern "C" void kernel_launch(void* const* d_in, const int* in_sizes, int n_in,
                              void* d_out, int out_size, void* d_ws,
                              size_t ws_size, hipStream_t stream) {
  const float* x  = (const float*)d_in[0];
  const int* pid  = (const int*)d_in[1];
  const unsigned char* am = (const unsigned char*)d_in[2];
  const float* Wq = (const float*)d_in[3];
  const float* Wk = (const float*)d_in[4];
  const float* Wv = (const float*)d_in[5];
  const float* Wo = (const float*)d_in[6];
  float* out = (float*)d_out;

  char* ws = (char*)d_ws;
  size_t off = 0;
  auto alloc = [&](size_t bytes) -> void* {
    void* p = ws + off;
    off += (bytes + 255) & ~(size_t)255;
    return p;
  };

  bf16* xb  = (bf16*)alloc((size_t)NSEQ * HID * 2);
  bf16* wqb = (bf16*)alloc((size_t)INNER * HID * 2);
  bf16* wkb = (bf16*)alloc((size_t)INNER * HID * 2);
  bf16* wvb = (bf16*)alloc((size_t)INNER * HID * 2);
  bf16* wob = (bf16*)alloc((size_t)HID * INNER * 2);
  bf16* qh  = (bf16*)alloc((size_t)HN * NSEQ * DD * 2);
  bf16* kh  = (bf16*)alloc((size_t)HN * NSEQ * DD * 2);
  bf16* vt  = (bf16*)alloc((size_t)HN * DD * NSEQ * 2);
  bf16* atb = (bf16*)alloc((size_t)NSEQ * INNER * 2);
  int*  sem = (int*)alloc((size_t)NSEQ * 4);
  int*  act = (int*)alloc((size_t)NSEQ * 4);

  // K0: conversions
  {
    int n = NSEQ * HID;
    k_cvt_bf16<<<(n + 255) / 256, 256, 0, stream>>>(x, xb, n);
    n = INNER * HID;
    k_cvt_bf16<<<(n + 255) / 256, 256, 0, stream>>>(Wq, wqb, n);
    k_cvt_bf16<<<(n + 255) / 256, 256, 0, stream>>>(Wk, wkb, n);
    k_cvt_bf16<<<(n + 255) / 256, 256, 0, stream>>>(Wv, wvb, n);
    n = HID * INNER;
    k_cvt_bf16<<<(n + 255) / 256, 256, 0, stream>>>(Wo, wob, n);
  }

  // K1: semantic position scan
  k_sem_scan<<<1, 1024, 0, stream>>>(am, sem, act);

  // K2: fused QKV + RoPE (TDM-staged B panel)
  k_qkv_rope<<<dim3(32, HN, 3), 128, 0, stream>>>(xb, wqb, wkb, wvb, pid, qh,
                                                  kh, vt);

  // K3: sliding-window flash attention (2048 waves, 4 per block)
  k_attn<<<(HN * 256) / WPB, 32 * WPB, 0, stream>>>(qh, kh, vt, sem, act, atb);

  // K4: output projection
  k_out_proj<<<dim3(32, 32), 128, 0, stream>>>(atb, wob, out);

  (void)in_sizes; (void)n_in; (void)out_size; (void)ws_size;
}
